// CrossAttentionDecoder_57037165691018
// MI455X (gfx1250) — compile-verified, matrix operand order checked by reading, not run on hardware
//
#include <hip/hip_runtime.h>
#include <hip/hip_bf16.h>

// ---------------------------------------------------------------------------
// CrossAttentionDecoder for MI455X (gfx1250, wave32, WMMA + TDM).
// Dense math via v_wmma_f32_16x16x32_f16; GEMM weight tiles staged into LDS
// by the Tensor Data Mover (tensor_load_to_lds, TENSORcnt-tracked, double-
// buffered), with hardware LDS padding for bank-conflict-free B fragments.
// ---------------------------------------------------------------------------

#define B_    4
#define NQ_   8192
#define NA_   64
#define KV_   128
#define HID_  256
#define HEADS_ 4
#define DH_   64
#define FFN_  1024

typedef __attribute__((ext_vector_type(16))) _Float16 v16h;
typedef __attribute__((ext_vector_type(8)))  _Float16 v8h;
typedef __attribute__((ext_vector_type(8)))  float    v8f;
typedef unsigned int u32x4 __attribute__((ext_vector_type(4)));
typedef int          i32x4 __attribute__((ext_vector_type(4)));
typedef int          i32x8 __attribute__((ext_vector_type(8)));

#if defined(__has_builtin)
#if __has_builtin(__builtin_amdgcn_tensor_load_to_lds) && \
    __has_builtin(__builtin_amdgcn_s_wait_tensorcnt)
#define USE_TDM 1
#endif
#endif

#define LDSTRIDE 40   // 32 K-halves + 4-DWORD pad per row (conflict-free reads)

// Load one 16x32 f16 WMMA operand fragment (A layout; B uses the mirrored
// layout from [N][K] row-major storage, which is the identical access pattern).
__device__ __forceinline__ v16h load_frag_g(const _Float16* __restrict__ base,
                                            int ld, int lane) {
  int row  = lane & 15;
  int koff = (lane >> 4) << 3;         // 0 or 8 halves
  const _Float16* p = base + (size_t)row * ld + koff;
  v8h lo = *(const v8h*)(p);
  v8h hi = *(const v8h*)(p + 16);
  v16h f;
#pragma unroll
  for (int i = 0; i < 8; ++i) { f[i] = lo[i]; f[i + 8] = hi[i]; }
  return f;
}

__device__ __forceinline__ float gelu_exact(float v) {
  return 0.5f * v * (1.0f + erff(v * 0.70710678118654752f));
}

#ifdef USE_TDM
// Issue a TDM load of a [64 rows x 32 halves] tile (row stride K halves) from
// global into LDS, with pad_interval=16DW / pad_amount=4DW so the LDS image
// has the padded LDSTRIDE layout. Wave-uniform; call from one wave only.
__device__ __forceinline__ void tdm_issue_tile(const _Float16* gsrc, int K,
                                               const _Float16* ldst) {
  u32x4 g0; i32x8 g1; i32x4 g2 = 0, g3 = 0;
  const unsigned long long ga = (unsigned long long)(size_t)gsrc;
  g0[0] = 1u;                                    // count=1, user descriptor
  g0[1] = (unsigned)(size_t)ldst;                // lds_addr (bytes)
  g0[2] = (unsigned)ga;                          // global_addr[31:0]
  g0[3] = (unsigned)(ga >> 32) | (2u << 30);     // global_addr[56:32] | type=2
  // data_size=2B, pad_enable, pad_interval=16DW (code 3), pad_amount=4DW (code 3)
  g1[0] = (int)((1u << 16) | (1u << 20) | (3u << 22) | (3u << 25));
  g1[1] = (int)((unsigned)K << 16);                    // tensor_dim0[15:0]
  g1[2] = (int)(((unsigned)K >> 16) | (64u << 16));    // tensor_dim0[31:16] | tensor_dim1[15:0]=64
  g1[3] = (int)(32u << 16);                            // tile_dim0 = 32
  g1[4] = 64;                                          // tile_dim1 = 64
  g1[5] = K;                                           // tensor_dim0_stride[31:0]
  g1[6] = 0; g1[7] = 0;
#if __clang_major__ >= 23
  i32x8 g4 = 0;
  __builtin_amdgcn_tensor_load_to_lds(g0, g1, g2, g3, g4, 0);
#else
  __builtin_amdgcn_tensor_load_to_lds(g0, g1, g2, g3, 0);
#endif
}
#endif

// ---------------------------------------------------------------------------
// Generic WMMA GEMM: C[M][N] = A_f16[M][K] @ Wt_f16[N][K]^T + bias[N]
// Block: 256 threads (8 waves). Tile: 128 rows x 64 cols. Wt tile staged in
// LDS by TDM (double-buffered) or cooperative loads as fallback.
// OUTMODE: 0 = f32 store, 1 = f16 store, 2 = gelu + f16 store,
//          3 = f16 store transposed for V^T layout [b][N][NA_] (rows = b*NA_+a)
// ---------------------------------------------------------------------------
template <int OUTMODE>
__global__ void gemm_wmma(const _Float16* __restrict__ A,
                          const _Float16* __restrict__ Wt,
                          const float* __restrict__ bias,
                          float* __restrict__ Cf,
                          _Float16* __restrict__ Ch,
                          int M, int N, int K) {
  __shared__ _Float16 lds[2 * 64 * LDSTRIDE];
  const int wave = threadIdx.x >> 5;
  const int lane = threadIdx.x & 31;
  const int m0 = blockIdx.x * 128 + wave * 16;
  const int n0 = blockIdx.y * 64;

  v8f acc[4];
#pragma unroll
  for (int t = 0; t < 4; ++t)
#pragma unroll
    for (int r = 0; r < 8; ++r) acc[t][r] = 0.0f;

  const int row  = lane & 15;
  const int koff = (lane >> 4) << 3;

#ifdef USE_TDM
  if (wave == 0) tdm_issue_tile(Wt + (size_t)n0 * K, K, &lds[0]);
  int buf = 0;
  for (int k0 = 0; k0 < K; k0 += 32) {
    if (wave == 0) {
      if (k0 + 32 < K) {
        tdm_issue_tile(Wt + (size_t)n0 * K + k0 + 32, K,
                       &lds[(buf ^ 1) * 64 * LDSTRIDE]);
        __builtin_amdgcn_s_wait_tensorcnt(1);   // current tile landed
      } else {
        __builtin_amdgcn_s_wait_tensorcnt(0);
      }
    }
    __syncthreads();
    const _Float16* tile = &lds[buf * 64 * LDSTRIDE];
    __builtin_prefetch(A + (size_t)m0 * K + k0 + 32, 0, 0);  // next A tile
    v16h a = load_frag_g(A + (size_t)m0 * K + k0, K, lane);
#pragma unroll
    for (int t = 0; t < 4; ++t) {
      const _Float16* p = tile + (t * 16 + row) * LDSTRIDE + koff;
      v8h lo = *(const v8h*)(p);
      v8h hi = *(const v8h*)(p + 16);
      v16h b;
#pragma unroll
      for (int i = 0; i < 8; ++i) { b[i] = lo[i]; b[i + 8] = hi[i]; }
      acc[t] = __builtin_amdgcn_wmma_f32_16x16x32_f16(
          false, a, false, b, (short)0, acc[t], false, false);
    }
    __syncthreads();   // all waves done with buf before TDM refills it
    buf ^= 1;
  }
#else
  const int ln = threadIdx.x >> 2;          // 0..63 : N row of tile
  const int lk = (threadIdx.x & 3) << 3;    // 0,8,16,24 : K halves
  for (int k0 = 0; k0 < K; k0 += 32) {
    __syncthreads();
    *(v8h*)&lds[ln * LDSTRIDE + lk] =
        *(const v8h*)(Wt + (size_t)(n0 + ln) * K + k0 + lk);
    __syncthreads();
    v16h a = load_frag_g(A + (size_t)m0 * K + k0, K, lane);
#pragma unroll
    for (int t = 0; t < 4; ++t) {
      const _Float16* p = &lds[(t * 16 + row) * LDSTRIDE + koff];
      v8h lo = *(const v8h*)(p);
      v8h hi = *(const v8h*)(p + 16);
      v16h b;
#pragma unroll
      for (int i = 0; i < 8; ++i) { b[i] = lo[i]; b[i + 8] = hi[i]; }
      acc[t] = __builtin_amdgcn_wmma_f32_16x16x32_f16(
          false, a, false, b, (short)0, acc[t], false, false);
    }
  }
#endif

  const int cg = lane >> 4, cl = lane & 15;
#pragma unroll
  for (int t = 0; t < 4; ++t) {
    const int col = n0 + t * 16 + cl;
    const float bv = bias[col];
#pragma unroll
    for (int r = 0; r < 8; ++r) {
      const int orow = m0 + r + 8 * cg;
      float v = acc[t][r] + bv;
      if (OUTMODE == 0) {
        Cf[(size_t)orow * N + col] = v;
      } else if (OUTMODE == 2) {
        Ch[(size_t)orow * N + col] = (_Float16)gelu_exact(v);
      } else if (OUTMODE == 3) {
        const int b = orow >> 6, a = orow & 63;     // rows = b*NA_ + a
        Ch[((size_t)b * N + col) * NA_ + a] = (_Float16)v;
      } else {
        Ch[(size_t)orow * N + col] = (_Float16)v;
      }
    }
  }
}

// ---------------------------------------------------------------------------
// Fused attention: per wave handles (16 q-rows, one head).
// scores = (Q K^T)/8 + bias ; softmax over NA_=64 via half-wave shuffles ;
// P repacked via LDS to A-fragments ; out = P V via WMMA on V^T.
// ---------------------------------------------------------------------------
__global__ void attn_kernel(const _Float16* __restrict__ Q,    // [B][NQ][HID]
                            const _Float16* __restrict__ Kf,   // [B][NA][HID]
                            const _Float16* __restrict__ Vt,   // [B][HID][NA]
                            const float* __restrict__ Bias,    // [B][H][NQ][NA]
                            _Float16* __restrict__ O) {        // [B][NQ][HID]
  __shared__ _Float16 P[8][16 * NA_];
  const int wave = threadIdx.x >> 5;
  const int lane = threadIdx.x & 31;
  const int h  = wave & 3;
  const int q0 = blockIdx.x * 32 + (wave >> 2) * 16;
  const int b  = blockIdx.y;

  const _Float16* Qb = Q  + ((size_t)b * NQ_ + q0) * HID_ + h * DH_;
  const _Float16* Kb = Kf + (size_t)b * NA_ * HID_ + h * DH_;
  const _Float16* Vb = Vt + ((size_t)b * HID_ + h * DH_) * NA_;
  const float*  biasp = Bias + (((size_t)b * HEADS_ + h) * NQ_ + q0) * NA_;

  v8f s[4];
#pragma unroll
  for (int t = 0; t < 4; ++t)
#pragma unroll
    for (int r = 0; r < 8; ++r) s[t][r] = 0.0f;

#pragma unroll
  for (int kk = 0; kk < DH_; kk += 32) {
    v16h a = load_frag_g(Qb + kk, HID_, lane);
#pragma unroll
    for (int t = 0; t < 4; ++t) {
      v16h bf = load_frag_g(Kb + (size_t)(t * 16) * HID_ + kk, HID_, lane);
      s[t] = __builtin_amdgcn_wmma_f32_16x16x32_f16(
          false, a, false, bf, (short)0, s[t], false, false);
    }
  }

  const int cg = lane >> 4, cl = lane & 15;
  const float scale = 0.125f;   // 1/sqrt(64)
#pragma unroll
  for (int r = 0; r < 8; ++r) {
    const int q = r + 8 * cg;   // row within 16-row tile
    float v0 = s[0][r] * scale + biasp[(size_t)q * NA_ +  0 + cl];
    float v1 = s[1][r] * scale + biasp[(size_t)q * NA_ + 16 + cl];
    float v2 = s[2][r] * scale + biasp[(size_t)q * NA_ + 32 + cl];
    float v3 = s[3][r] * scale + biasp[(size_t)q * NA_ + 48 + cl];
    float m = fmaxf(fmaxf(v0, v1), fmaxf(v2, v3));
#pragma unroll
    for (int off = 8; off >= 1; off >>= 1) m = fmaxf(m, __shfl_xor(m, off, 16));
    float e0 = __expf(v0 - m), e1 = __expf(v1 - m);
    float e2 = __expf(v2 - m), e3 = __expf(v3 - m);
    float sum = e0 + e1 + e2 + e3;
#pragma unroll
    for (int off = 8; off >= 1; off >>= 1) sum += __shfl_xor(sum, off, 16);
    const float inv = 1.0f / sum;
    P[wave][q * NA_ +  0 + cl] = (_Float16)(e0 * inv);
    P[wave][q * NA_ + 16 + cl] = (_Float16)(e1 * inv);
    P[wave][q * NA_ + 32 + cl] = (_Float16)(e2 * inv);
    P[wave][q * NA_ + 48 + cl] = (_Float16)(e3 * inv);
  }

  v8f o[4];
#pragma unroll
  for (int t = 0; t < 4; ++t)
#pragma unroll
    for (int r = 0; r < 8; ++r) o[t][r] = 0.0f;

#pragma unroll
  for (int s0 = 0; s0 < NA_; s0 += 32) {
    // A fragment of P from LDS (within-wave store->load; DScnt handled by compiler)
    int rw   = lane & 15;
    int ko   = (lane >> 4) << 3;
    const _Float16* pp = &P[wave][rw * NA_ + s0 + ko];
    v8h lo = *(const v8h*)(pp);
    v8h hi = *(const v8h*)(pp + 16);
    v16h a;
#pragma unroll
    for (int i = 0; i < 8; ++i) { a[i] = lo[i]; a[i + 8] = hi[i]; }
#pragma unroll
    for (int t = 0; t < 4; ++t) {
      v16h bf = load_frag_g(Vb + (size_t)(t * 16) * NA_ + s0, NA_, lane);
      o[t] = __builtin_amdgcn_wmma_f32_16x16x32_f16(
          false, a, false, bf, (short)0, o[t], false, false);
    }
  }

  _Float16* Ob = O + ((size_t)b * NQ_ + q0) * HID_ + h * DH_;
#pragma unroll
  for (int t = 0; t < 4; ++t)
#pragma unroll
    for (int r = 0; r < 8; ++r)
      Ob[(size_t)(r + 8 * cg) * HID_ + t * 16 + cl] = (_Float16)o[t][r];
}

// ---------------------------------------------------------------------------
// residual + LayerNorm: one wave per 256-elem row (8 elems/lane).
// ---------------------------------------------------------------------------
__global__ void add_ln(const float* X, const float* __restrict__ Y,
                       const float* __restrict__ G, const float* __restrict__ Bt,
                       float* Xo, _Float16* __restrict__ Xh) {
  const int wave = threadIdx.x >> 5, lane = threadIdx.x & 31;
  const size_t row = (size_t)blockIdx.x * 8 + wave;
  const float* xr = X + row * HID_;
  const float* yr = Y + row * HID_;
  float v[8]; float sum = 0.0f;
#pragma unroll
  for (int i = 0; i < 8; ++i) { v[i] = xr[lane + i * 32] + yr[lane + i * 32]; sum += v[i]; }
#pragma unroll
  for (int off = 16; off >= 1; off >>= 1) sum += __shfl_xor(sum, off, 32);
  const float mean = sum * (1.0f / HID_);
  float var = 0.0f;
#pragma unroll
  for (int i = 0; i < 8; ++i) { float d = v[i] - mean; var += d * d; }
#pragma unroll
  for (int off = 16; off >= 1; off >>= 1) var += __shfl_xor(var, off, 32);
  const float rstd = rsqrtf(var * (1.0f / HID_) + 1e-5f);
#pragma unroll
  for (int i = 0; i < 8; ++i) {
    const int c = lane + i * 32;
    const float o = (v[i] - mean) * rstd * G[c] + Bt[c];
    Xo[row * HID_ + c] = o;
    Xh[row * HID_ + c] = (_Float16)o;
  }
}

// x = query_pos @ Wqp + bqp (K=3): one block per row, thread = col.
__global__ void init_x(const float* __restrict__ QP, const float* __restrict__ W,
                       const float* __restrict__ Bv, float* __restrict__ Xo,
                       _Float16* __restrict__ Xh) {
  const size_t row = blockIdx.x; const int c = threadIdx.x;
  const float* p = QP + row * 3;
  const float o = Bv[c] + p[0] * W[c] + p[1] * W[HID_ + c] + p[2] * W[2 * HID_ + c];
  Xo[row * HID_ + c] = o;
  Xh[row * HID_ + c] = (_Float16)o;
}

// atom_enc = atom_features @ Wkv + bkv + fourier(atom_pos) @ Wap + bap
__global__ void atom_enc_k(const float* __restrict__ AF, const float* __restrict__ AP,
                           const float* __restrict__ KW, const float* __restrict__ KB,
                           const float* __restrict__ PW, const float* __restrict__ PB,
                           float* __restrict__ AE, _Float16* __restrict__ AEh) {
  const int row = blockIdx.x; const int c = threadIdx.x;
  const float* af = AF + (size_t)row * KV_;
  float acc = KB[c] + PB[c];
#pragma unroll 4
  for (int i = 0; i < KV_; ++i) acc += af[i] * KW[(size_t)i * HID_ + c];
  const float* ap = AP + (size_t)row * 3;
  float e[27];
  e[0] = ap[0]; e[1] = ap[1]; e[2] = ap[2];
#pragma unroll
  for (int d = 0; d < 3; ++d) {
    float fr = 3.14159265358979323846f;
#pragma unroll
    for (int f = 0; f < 4; ++f) {
      const float s = ap[d] * fr;
      e[3 + d * 8 + f]     = __sinf(s);
      e[3 + d * 8 + 4 + f] = __cosf(s);
      fr *= 2.0f;
    }
  }
#pragma unroll
  for (int i = 0; i < 27; ++i) acc += e[i] * PW[(size_t)i * HID_ + c];
  AE[(size_t)row * HID_ + c] = acc;
  AEh[(size_t)row * HID_ + c] = (_Float16)acc;
}

// bias[b][h][q][a] = softplus(-dist(q,a)*w[h] + b[h]); shared by both layers.
__global__ void bias_k(const float* __restrict__ QP, const float* __restrict__ AP,
                       const float* __restrict__ DW, const float* __restrict__ DB,
                       float* __restrict__ Bias) {
  const int t = threadIdx.x; const int a = t & 63; const int qs = t >> 6;
  const size_t bq = (size_t)blockIdx.x * 4 + qs;       // over B*NQ
  const int b = (int)(bq / NQ_); const int q = (int)(bq % NQ_);
  const float dx = QP[bq * 3 + 0] - AP[((size_t)b * NA_ + a) * 3 + 0];
  const float dy = QP[bq * 3 + 1] - AP[((size_t)b * NA_ + a) * 3 + 1];
  const float dz = QP[bq * 3 + 2] - AP[((size_t)b * NA_ + a) * 3 + 2];
  const float dist = sqrtf(dx * dx + dy * dy + dz * dz);
#pragma unroll
  for (int h = 0; h < HEADS_; ++h) {
    const float z = -dist * DW[h] + DB[h];
    const float sp = (z > 20.0f) ? z : log1pf(__expf(z));
    Bias[(((size_t)b * HEADS_ + h) * NQ_ + q) * NA_ + a] = sp;
  }
}

// f32 [din][dout] -> f16 transposed [dout][din] (WMMA B-operand layout)
__global__ void wconv(const float* __restrict__ S, _Float16* __restrict__ D,
                      int din, int dout) {
  const size_t i = (size_t)blockIdx.x * 256 + threadIdx.x;
  if (i >= (size_t)din * dout) return;
  const size_t o = i % dout, in_ = i / dout;
  D[o * din + in_] = (_Float16)S[i];
}

// ---------------------------------------------------------------------------
extern "C" void kernel_launch(void* const* d_in, const int* in_sizes, int n_in,
                              void* d_out, int out_size, void* d_ws, size_t ws_size,
                              hipStream_t stream) {
  (void)in_sizes; (void)n_in; (void)out_size; (void)ws_size;
  // Input order: DFS insertion order of setup_inputs() dict.
  const float* query_pos = (const float*)d_in[0];
  const float* atom_feat = (const float*)d_in[1];
  const float* atom_pos  = (const float*)d_in[2];
  const float* qp_w = (const float*)d_in[3];
  const float* qp_b = (const float*)d_in[4];
  const float* kv_w = (const float*)d_in[5];
  const float* kv_b = (const float*)d_in[6];
  const float* ap_w = (const float*)d_in[7];
  const float* ap_b = (const float*)d_in[8];
  const float* dp_w = (const float*)d_in[9];
  const float* dp_b = (const float*)d_in[10];

  float* xout = (float*)d_out;                 // f32 master activations

  // workspace carve-up
  char* ws = (char*)d_ws;
  size_t off = 0;
  auto take = [&](size_t bytes) -> void* {
    void* p = ws + off; off = (off + bytes + 255) & ~(size_t)255; return p;
  };
  const size_t NTOK = (size_t)B_ * NQ_;        // 32768
  _Float16* wT[2][6];
  const size_t wsz[6] = {HID_*HID_, HID_*HID_, HID_*HID_, HID_*HID_,
                         (size_t)HID_*FFN_, (size_t)FFN_*HID_};
  for (int l = 0; l < 2; ++l)
    for (int j = 0; j < 6; ++j) wT[l][j] = (_Float16*)take(wsz[j] * 2);
  _Float16* xf16   = (_Float16*)take(NTOK * HID_ * 2);
  _Float16* qf16   = (_Float16*)take(NTOK * HID_ * 2);
  _Float16* attnh  = (_Float16*)take(NTOK * HID_ * 2);
  _Float16* hf16   = (_Float16*)take(NTOK * FFN_ * 2);
  float*    yf32   = (float*)take(NTOK * HID_ * 4);
  float*    biasf  = (float*)take((size_t)B_ * HEADS_ * NQ_ * NA_ * 4);
  float*    aef32  = (float*)take((size_t)B_ * NA_ * HID_ * 4);
  _Float16* aef16  = (_Float16*)take((size_t)B_ * NA_ * HID_ * 2);
  _Float16* kf16   = (_Float16*)take((size_t)B_ * NA_ * HID_ * 2);
  _Float16* vTf16  = (_Float16*)take((size_t)B_ * HID_ * NA_ * 2);

  // weight conversion (per-layer W matrices, f16 transposed)
  for (int l = 0; l < 2; ++l) {
    const int base = 11 + 16 * l;
    wconv<<<256, 256, 0, stream>>>((const float*)d_in[base + 0], wT[l][0], HID_, HID_);
    wconv<<<256, 256, 0, stream>>>((const float*)d_in[base + 2], wT[l][1], HID_, HID_);
    wconv<<<256, 256, 0, stream>>>((const float*)d_in[base + 4], wT[l][2], HID_, HID_);
    wconv<<<256, 256, 0, stream>>>((const float*)d_in[base + 6], wT[l][3], HID_, HID_);
    wconv<<<1024, 256, 0, stream>>>((const float*)d_in[base + 8], wT[l][4], HID_, FFN_);
    wconv<<<1024, 256, 0, stream>>>((const float*)d_in[base + 10], wT[l][5], FFN_, HID_);
  }

  init_x<<<NTOK, HID_, 0, stream>>>(query_pos, qp_w, qp_b, xout, xf16);
  atom_enc_k<<<B_ * NA_, HID_, 0, stream>>>(atom_feat, atom_pos, kv_w, kv_b,
                                            ap_w, ap_b, aef32, aef16);
  bias_k<<<(B_ * NQ_) / 4, 256, 0, stream>>>(query_pos, atom_pos, dp_w, dp_b, biasf);

  const dim3 gBig(NTOK / 128, HID_ / 64);        // (256, 4)
  const dim3 gFfn1(NTOK / 128, FFN_ / 64);       // (256, 16)
  const dim3 gKV((B_ * NA_) / 128, HID_ / 64);   // (2, 4)

  for (int l = 0; l < 2; ++l) {
    const int base = 11 + 16 * l;
    const float* bq  = (const float*)d_in[base + 1];
    const float* bk  = (const float*)d_in[base + 3];
    const float* bv  = (const float*)d_in[base + 5];
    const float* bo  = (const float*)d_in[base + 7];
    const float* b1  = (const float*)d_in[base + 9];
    const float* b2  = (const float*)d_in[base + 11];
    const float* g1  = (const float*)d_in[base + 12];
    const float* gb1 = (const float*)d_in[base + 13];
    const float* g2  = (const float*)d_in[base + 14];
    const float* gb2 = (const float*)d_in[base + 15];

    gemm_wmma<1><<<gBig, 256, 0, stream>>>(xf16, wT[l][0], bq, nullptr, qf16,
                                           (int)NTOK, HID_, HID_);
    gemm_wmma<1><<<gKV, 256, 0, stream>>>(aef16, wT[l][1], bk, nullptr, kf16,
                                          B_ * NA_, HID_, HID_);
    gemm_wmma<3><<<gKV, 256, 0, stream>>>(aef16, wT[l][2], bv, nullptr, vTf16,
                                          B_ * NA_, HID_, HID_);
    attn_kernel<<<dim3(NQ_ / 32, B_), 256, 0, stream>>>(qf16, kf16, vTf16,
                                                        biasf, attnh);
    gemm_wmma<0><<<gBig, 256, 0, stream>>>(attnh, wT[l][3], bo, yf32, nullptr,
                                           (int)NTOK, HID_, HID_);
    add_ln<<<NTOK / 8, 256, 0, stream>>>(xout, yf32, g1, gb1, xout, xf16);
    gemm_wmma<2><<<gFfn1, 256, 0, stream>>>(xf16, wT[l][4], b1, nullptr, hf16,
                                            (int)NTOK, FFN_, HID_);
    gemm_wmma<0><<<gBig, 256, 0, stream>>>(hf16, wT[l][5], b2, yf32, nullptr,
                                           (int)NTOK, HID_, FFN_);
    add_ln<<<NTOK / 8, 256, 0, stream>>>(xout, yf32, g2, gb2, xout, xf16);
  }
}
